// QLSTM_65481071396923
// MI455X (gfx1250) — matrix operand order repaired and checked by reading, not compiled, not run on hardware
//
#include <hip/hip_runtime.h>

typedef __attribute__((ext_vector_type(16))) __bf16 v16bf;
typedef __attribute__((ext_vector_type(8)))  float  v8f;

#define SEQ   256
#define BATCH 512
#define DIN   264   // input_dim + n_qubits
#define IN    256   // input_dim
#define NQ    8     // n_qubits == hidden
#define NCOL  32    // 4 gates * 8 qubits
#define MROWS (SEQ * BATCH)  // 131072

// ---------------------------------------------------------------------------
// Kernel 1: pack the x-part of Wf/Wi/Wu/Wo (each (8,264) row-major, cols 0..255)
// into a WMMA bf16 B-operand layout in contiguous k order (same as the A tile):
//   flat index = ((kc*2 + nt)*32 + lane)*16 + h
//   lanes 0-15 hold K = kc*32 + h      (h = 0..15)
//   lanes 16-31 hold K = kc*32 + 16 + h
//   N = nt*16 + (lane & 15),  N -> (gate, qubit) = (N>>3, N&7)
// ---------------------------------------------------------------------------
__global__ __launch_bounds__(256) void qlstm_pack_w(
    const float* __restrict__ Wf, const float* __restrict__ Wi,
    const float* __restrict__ Wu, const float* __restrict__ Wo,
    __bf16* __restrict__ bpack) {
  int idx  = blockIdx.x * blockDim.x + threadIdx.x;  // 0..8191
  int h    = idx & 15;
  int lane = (idx >> 4) & 31;
  int tile = idx >> 9;          // kc*2 + nt, 0..15
  int nt   = tile & 1;
  int kc   = tile >> 1;
  int K    = kc * 32 + ((lane >= 16) ? 16 : 0) + h;
  int N    = nt * 16 + (lane & 15);
  int g    = N >> 3;
  int q    = N & 7;
  const float* W = (g == 0) ? Wf : (g == 1) ? Wi : (g == 2) ? Wu : Wo;
  bpack[idx] = (__bf16)W[q * DIN + K];
}

// ---------------------------------------------------------------------------
// Kernel 2: Z[m][n] = x[m] @ Wx[n],  m in [0,131072), n in [0,32)
// One wave per 16-row tile; two 16x16 accumulators cover all 32 columns.
// A tile (16x32 bf16) built in-register from f32 global loads following the
// contiguous-k 16-bit A layout (same k order as the B pack):
//   lanes 0-15 : halves 0-15 = K 0-15
//   lanes 16-31: halves 0-15 = K 16-31
// ---------------------------------------------------------------------------
__global__ __launch_bounds__(256) void qlstm_gemm(
    const float* __restrict__ x, const __bf16* __restrict__ bpack,
    float* __restrict__ Z) {
  int lane = threadIdx.x & 31;
  int gw   = (blockIdx.x * blockDim.x + threadIdx.x) >> 5;  // 0..8191
  const float*  arow = x + (size_t)(gw * 16 + (lane & 15)) * IN;
  const int     sel  = (lane >= 16) ? 16 : 0;
  const v16bf*  bp   = (const v16bf*)bpack;

  v8f acc0 = {};
  v8f acc1 = {};
#pragma unroll
  for (int kc = 0; kc < 8; ++kc) {
    const float4* p0 = (const float4*)(arow + kc * 32 + sel);
    const float4* p1 = (const float4*)(arow + kc * 32 + 8 + sel);
    float4 f0 = p0[0], f1 = p0[1];
    float4 f2 = p1[0], f3 = p1[1];
    v16bf a;
    a[0]  = (__bf16)f0.x; a[1]  = (__bf16)f0.y; a[2]  = (__bf16)f0.z; a[3]  = (__bf16)f0.w;
    a[4]  = (__bf16)f1.x; a[5]  = (__bf16)f1.y; a[6]  = (__bf16)f1.z; a[7]  = (__bf16)f1.w;
    a[8]  = (__bf16)f2.x; a[9]  = (__bf16)f2.y; a[10] = (__bf16)f2.z; a[11] = (__bf16)f2.w;
    a[12] = (__bf16)f3.x; a[13] = (__bf16)f3.y; a[14] = (__bf16)f3.z; a[15] = (__bf16)f3.w;

    v16bf b0 = bp[(kc * 2 + 0) * 32 + lane];
    v16bf b1 = bp[(kc * 2 + 1) * 32 + lane];

    acc0 = __builtin_amdgcn_wmma_f32_16x16x32_bf16(false, a, false, b0,
                                                   (short)0, acc0, false, false);
    acc1 = __builtin_amdgcn_wmma_f32_16x16x32_bf16(false, a, false, b1,
                                                   (short)0, acc1, false, false);
  }

  // C/D layout: VGPR r -> row r (lanes 0-15) / row r+8 (lanes 16-31), col = lane%16
  int rbase = gw * 16 + ((lane >= 16) ? 8 : 0);
  int col   = lane & 15;
#pragma unroll
  for (int r = 0; r < 8; ++r) {
    float* zr    = Z + (size_t)(rbase + r) * NCOL;
    zr[col]      = acc0[r];
    zr[col + 16] = acc1[r];
  }
}

// ---------------------------------------------------------------------------
// Kernel 3: sequential LSTM recurrence. One wave per batch row.
// lane = g*8 + q  (g = gate: 0=f,1=i,2=u,3=o ; q = qubit)
// ---------------------------------------------------------------------------
__global__ __launch_bounds__(256) void qlstm_recur(
    const float* __restrict__ Z,
    const float* __restrict__ Wf, const float* __restrict__ Wi,
    const float* __restrict__ Wu, const float* __restrict__ Wo,
    const float* __restrict__ bf_, const float* __restrict__ bi_,
    const float* __restrict__ bu_, const float* __restrict__ bo_,
    const float* __restrict__ pf,  const float* __restrict__ pi_,
    const float* __restrict__ pu,  const float* __restrict__ po,
    float* __restrict__ out) {
  int lane = threadIdx.x & 31;
  int b    = (blockIdx.x * blockDim.x + threadIdx.x) >> 5;  // 0..511
  int g    = lane >> 3;
  int q    = lane & 7;

  const float* W  = (g == 0) ? Wf  : (g == 1) ? Wi  : (g == 2) ? Wu  : Wo;
  const float* bv = (g == 0) ? bf_ : (g == 1) ? bi_ : (g == 2) ? bu_ : bo_;
  const float* pv = (g == 0) ? pf  : (g == 1) ? pi_ : (g == 2) ? pu  : po;

  float wh[NQ];
#pragma unroll
  for (int j = 0; j < NQ; ++j) wh[j] = W[q * DIN + IN + j];
  float bias = bv[q];
  float ct   = cosf(pv[q]);

  float hx[NQ];
#pragma unroll
  for (int j = 0; j < NQ; ++j) hx[j] = 0.f;
  float cxq = 0.f, hown = 0.f;

  float* stacked = out;                         // (SEQ, BATCH, NQ)
  float* hx_out  = out + (size_t)SEQ * BATCH * NQ;
  float* cx_out  = hx_out + (size_t)BATCH * NQ;

  for (int t = 0; t < SEQ; ++t) {
    // pre-activation: precomputed x-part + bias + recurrent 8-dim dot
    float z = bias + Z[((size_t)t * BATCH + b) * NCOL + lane];
#pragma unroll
    for (int j = 0; j < NQ; ++j) z = fmaf(wh[j], hx[j], z);

    // qgate: cumprod over the 8-lane segment of cos(z)*cos(theta)
    float c = cosf(z) * ct;
#pragma unroll
    for (int off = 1; off < 8; off <<= 1) {
      float other = __shfl_up(c, (unsigned)off, 8);
      if (q >= off) c *= other;
    }

    // activations: f,i,o -> sigmoid ; u -> tanh
    float val = (g == 2) ? tanhf(c) : (1.f / (1.f + __expf(-c)));

    // gather the four gate values for this qubit
    float fq = __shfl(val, q, 32);
    float iq = __shfl(val, q + 8, 32);
    float uq = __shfl(val, q + 16, 32);
    float oq = __shfl(val, q + 24, 32);

    cxq      = fmaf(fq, cxq, iq * uq);
    float hq = oq * tanhf(cxq);

    if (g == 0) stacked[(size_t)t * BATCH * NQ + b * NQ + q] = hq;
    hown = hq;

    // rebroadcast h to all lanes for next step's dot product
#pragma unroll
    for (int j = 0; j < NQ; ++j) hx[j] = __shfl(hq, j, 32);
  }

  if (g == 0) {
    hx_out[b * NQ + q] = hown;
    cx_out[b * NQ + q] = cxq;
  }
}

// ---------------------------------------------------------------------------
extern "C" void kernel_launch(void* const* d_in, const int* in_sizes, int n_in,
                              void* d_out, int out_size, void* d_ws, size_t ws_size,
                              hipStream_t stream) {
  const float* x   = (const float*)d_in[0];
  const float* Wf  = (const float*)d_in[1];
  const float* bf_ = (const float*)d_in[2];
  const float* Wi  = (const float*)d_in[3];
  const float* bi_ = (const float*)d_in[4];
  const float* Wu  = (const float*)d_in[5];
  const float* bu_ = (const float*)d_in[6];
  const float* Wo  = (const float*)d_in[7];
  const float* bo_ = (const float*)d_in[8];
  const float* pf  = (const float*)d_in[9];
  const float* pi_ = (const float*)d_in[10];
  const float* pu  = (const float*)d_in[11];
  const float* po  = (const float*)d_in[12];

  // workspace layout: [0,16KB) packed bf16 weights ; [16KB, +16.8MB) Z (f32)
  __bf16* bpack = (__bf16*)d_ws;
  float*  Z     = (float*)((char*)d_ws + 16384);

  // 1) pack weights: 8192 elements
  qlstm_pack_w<<<32, 256, 0, stream>>>(Wf, Wi, Wu, Wo, bpack);
  // 2) big GEMM: 8192 waves (16-row tiles) -> 1024 blocks x 256 threads
  qlstm_gemm<<<1024, 256, 0, stream>>>(x, bpack, Z);
  // 3) recurrence: 512 waves -> 64 blocks x 256 threads
  qlstm_recur<<<64, 256, 0, stream>>>(Z, Wf, Wi, Wu, Wo,
                                      bf_, bi_, bu_, bo_,
                                      pf, pi_, pu, po, (float*)d_out);
}
